// MARQuantizer2DEMA_3779571220595
// MI455X (gfx1250) — matrix operand order repaired and checked by reading, not compiled, not run on hardware
//
#include <hip/hip_runtime.h>
#include <math.h>

typedef _Float16 f16;
typedef __attribute__((ext_vector_type(16))) _Float16 v16h;
typedef __attribute__((ext_vector_type(8)))  float    v8f;

#define B_ 8
#define C_ 256
#define J_ 16
#define L_ 256
#define NB 8192
#define K9 2304                    // 9 * C
#define NUMEL (B_*C_*J_*L_)        // 8388608
#define NIDX 43296                 // 32+256+2048+8192+32768

// ---------------------------------------------------------------- init / prep
__global__ void k_init(const float* __restrict__ f, float* __restrict__ f_rest,
                       float* __restrict__ f_hat) {
    int i = blockIdx.x * blockDim.x + threadIdx.x;
    if (i < NUMEL) { f_rest[i] = f[i]; f_hat[i] = 0.0f; }
}

__global__ void k_init_small(float* __restrict__ counts, float* __restrict__ loss_acc) {
    int i = blockIdx.x * blockDim.x + threadIdx.x;
    if (i < NB) counts[i] = 0.0f;
    if (i < 8)  loss_acc[i] = 0.0f;
}

__global__ void k_prep_cb(const float* __restrict__ cb, f16* __restrict__ cb16,
                          float* __restrict__ cb_sq) {
    int i = blockIdx.x * blockDim.x + threadIdx.x;
    if (i < NB * C_) cb16[i] = (f16)cb[i];
    if (i < NB) {
        const float* row = cb + (size_t)i * C_;
        float s = 0.0f;
        for (int c = 0; c < C_; ++c) s += row[c] * row[c];
        cb_sq[i] = s;
    }
}

// phi_w [4][cout][cin][3][3] -> w16 [4][cout][tap*256 + cin], tap = ky*3+kx
__global__ void k_prep_w(const float* __restrict__ pw, f16* __restrict__ w16) {
    int i = blockIdx.x * blockDim.x + threadIdx.x;
    if (i >= 4 * C_ * K9) return;
    int k = i % K9;
    int o = (i / K9) % C_;
    int p = i / (K9 * C_);
    int tap = k / C_, ci = k % C_;
    w16[i] = (f16)pw[(((size_t)(p * C_ + o) * C_ + ci) * 9) + tap];
}

// -------------------------------------------------- area mean-pool -> f16 rows
__global__ void k_area(const float* __restrict__ f_rest, f16* __restrict__ r16,
                       int sn, int tn) {
    int i = blockIdx.x * blockDim.x + threadIdx.x;     // N*C threads
    int N = B_ * sn * tn;
    if (i >= N * C_) return;
    int c   = i % C_;
    int row = i / C_;
    int tl  = row % tn;
    int sj  = (row / tn) % sn;
    int b   = row / (tn * sn);
    int bj = J_ / sn, bl = L_ / tn;
    const float* base = f_rest + (((size_t)(b * C_ + c) * J_ + sj * bj) * L_ + tl * bl);
    float s = 0.0f;
    for (int y = 0; y < bj; ++y)
        for (int x = 0; x < bl; ++x) s += base[y * L_ + x];
    r16[(size_t)row * C_ + c] = (f16)(s / (float)(bj * bl));
}

// ---------------------------------------------- WMMA codebook distance argmin
// One block = 16 rest rows vs all 8192 codes. 8 waves stripe the 512 code tiles.
__global__ void __launch_bounds__(256) k_argmin(const f16* __restrict__ r16,
                                                const f16* __restrict__ cb16,
                                                const float* __restrict__ cb_sq,
                                                int* __restrict__ idx_out) {
    __shared__ f16  lds_a[16 * 256];
    __shared__ float s_min[8][16];
    __shared__ int   s_idx[8][16];

    int tid  = threadIdx.x;
    int lane = tid & 31;
    int w    = tid >> 5;
    int tile = blockIdx.x;

    { // stage the 16x256 A tile into LDS
        const f16* src = r16 + (size_t)tile * 16 * 256;
        for (int e = tid; e < 16 * 256; e += 256) lds_a[e] = src[e];
    }
    __syncthreads();

    int half = lane >> 4, m = lane & 15;

    // Build all 8 K-step A fragments (contiguous k = 16*half + s, matching B)
    v16h a[8];
#pragma unroll
    for (int ks = 0; ks < 8; ++ks) {
        const f16* rowp = lds_a + m * 256 + ks * 32;
#pragma unroll
        for (int i = 0; i < 8; ++i) { a[ks][i]     = rowp[16 * half + i];
                                      a[ks][8 + i] = rowp[16 * half + 8 + i]; }
    }

    float best[8]; int bidx[8];
#pragma unroll
    for (int r = 0; r < 8; ++r) { best[r] = 3.4e38f; bidx[r] = 0x7fffffff; }

    for (int ct = w; ct < NB / 16; ct += 8) {
        int code = ct * 16 + m;                       // this lane's column N
        v8f acc = {};
#pragma unroll
        for (int ks = 0; ks < 8; ++ks) {
            const f16* bp = cb16 + (size_t)code * 256 + ks * 32 + 16 * half;
            v16h bf;
#pragma unroll
            for (int i = 0; i < 16; ++i) bf[i] = bp[i];
            acc = __builtin_amdgcn_wmma_f32_16x16x32_f16(false, a[ks], false, bf,
                                                         (short)0, acc, false, false);
        }
        float csq = cb_sq[code];
#pragma unroll
        for (int r = 0; r < 8; ++r) {
            float s = csq - 2.0f * acc[r];            // row M = r + 8*half
            if (s < best[r] || (s == best[r] && code < bidx[r])) { best[r] = s; bidx[r] = code; }
        }
    }

    // reduce across the 16 lanes of each half (masks 1..8 stay in-group)
#pragma unroll
    for (int r = 0; r < 8; ++r) {
        float s = best[r]; int ix = bidx[r];
#pragma unroll
        for (int off = 8; off >= 1; off >>= 1) {
            float os = __shfl_xor(s, off, 32);
            int   oi = __shfl_xor(ix, off, 32);
            if (os < s || (os == s && oi < ix)) { s = os; ix = oi; }
        }
        if (m == 0) { s_min[w][r + 8 * half] = s; s_idx[w][r + 8 * half] = ix; }
    }
    __syncthreads();
    if (tid < 16) {
        float s = s_min[0][tid]; int ix = s_idx[0][tid];
        for (int ww = 1; ww < 8; ++ww) {
            float os = s_min[ww][tid]; int oi = s_idx[ww][tid];
            if (os < s || (os == s && oi < ix)) { s = os; ix = oi; }
        }
        idx_out[tile * 16 + tid] = ix;
    }
}

// ---------------------------------------------------- gather + bicubic upsample
__device__ inline float cubw(float x) {
    const float A = -0.75f;
    float ax = fabsf(x);
    if (ax <= 1.0f) return ((A + 2.0f) * ax - (A + 3.0f)) * ax * ax + 1.0f;
    if (ax < 2.0f)  return A * (ax * ((ax - 5.0f) * ax + 8.0f) - 4.0f);
    return 0.0f;
}

// writes h in NHWC f16 so conv B-fragments are contiguous 32B f16 loads
__global__ void k_upsample(const float* __restrict__ cb, const int* __restrict__ idx,
                           f16* __restrict__ h16, int sn, int tn) {
    int gid = blockIdx.x * blockDim.x + threadIdx.x;
    if (gid >= NUMEL) return;
    int c = gid & 255, l = (gid >> 8) & 255, j = (gid >> 16) & 15, b = gid >> 20;
    float val;
    if (sn == J_ && tn == L_) {
        val = cb[(size_t)idx[(b * sn + j) * tn + l] * C_ + c];
    } else {
        float sy = (float)sn / J_, sx = (float)tn / L_;
        float cy = (j + 0.5f) * sy - 0.5f;  float fy = floorf(cy); float ty = cy - fy;
        float cx = (l + 0.5f) * sx - 0.5f;  float fx = floorf(cx); float tx = cx - fx;
        int iy[4], ix[4]; float wy[4], wx[4];
#pragma unroll
        for (int o = 0; o < 4; ++o) {
            int yy = (int)fy + o - 1; iy[o] = min(max(yy, 0), sn - 1); wy[o] = cubw(ty - (float)(o - 1));
            int xx = (int)fx + o - 1; ix[o] = min(max(xx, 0), tn - 1); wx[o] = cubw(tx - (float)(o - 1));
        }
        val = 0.0f;
#pragma unroll
        for (int oy = 0; oy < 4; ++oy) {
            float acc = 0.0f;
#pragma unroll
            for (int ox = 0; ox < 4; ++ox) {
                int row = (b * sn + iy[oy]) * tn + ix[ox];
                acc += wx[ox] * cb[(size_t)idx[row] * C_ + c];
            }
            val += wy[oy] * acc;
        }
    }
    h16[gid] = (f16)val;   // gid == ((b*16+j)*256+l)*256+c
}

// ------------------------------- implicit-GEMM 3x3 conv + phi blend + updates
// Block: 32-pixel strip (contiguous l, two 16-wide N tiles). 8 waves x
// 2 cout-tiles x 2 n-tiles = 4 WMMAs/K-step; A-fragments (weights) are reused
// across both N tiles -> halves weight traffic from L2.
__global__ void __launch_bounds__(256) k_conv(const f16* __restrict__ h16,
                                              const f16* __restrict__ w16,
                                              const float* __restrict__ pb,
                                              const float* __restrict__ f,
                                              float* __restrict__ f_hat,
                                              float* __restrict__ f_rest,
                                              float* __restrict__ loss_acc, int si) {
    __shared__ float s_loss[8];
    int tid  = threadIdx.x;
    int lane = tid & 31;
    int w    = tid >> 5;
    int half = lane >> 4, m = lane & 15;

    int t  = blockIdx.x;                 // 0..1023
    int b  = t >> 6;
    int rm = t & 63;
    int j  = rm >> 3;
    int l0 = (rm & 7) << 5;              // 32-pixel strip
    int la = l0 + m;                     // N tile 0 column
    int lb = l0 + 16 + m;                // N tile 1 column

    int cout0 = (w * 2) * 16, cout1 = (w * 2 + 1) * 16;
    const f16* wr0 = w16 + (size_t)(cout0 + m) * K9;
    const f16* wr1 = w16 + (size_t)(cout1 + m) * K9;

    v8f acc00 = {}, acc01 = {}, acc10 = {}, acc11 = {};
    for (int kk = 0; kk < 72; ++kk) {    // K = 2304 in steps of 32
        int tap = kk >> 3;
        int ci0 = (kk & 7) << 5;
        int dj  = tap / 3 - 1;
        int dl  = tap % 3 - 1;
        int jj  = j + dj;
        bool jok = (jj >= 0 && jj < J_);

        v16h bf0, bf1;
        {
            int ll = la + dl;
            if (jok && ll >= 0 && ll < L_) {
                const f16* q = h16 + (((size_t)(b * J_ + jj) * L_ + ll) * C_ + ci0 + 16 * half);
#pragma unroll
                for (int i = 0; i < 16; ++i) bf0[i] = q[i];
            } else {
#pragma unroll
                for (int i = 0; i < 16; ++i) bf0[i] = (f16)0.0f;
            }
        }
        {
            int ll = lb + dl;
            if (jok && ll >= 0 && ll < L_) {
                const f16* q = h16 + (((size_t)(b * J_ + jj) * L_ + ll) * C_ + ci0 + 16 * half);
#pragma unroll
                for (int i = 0; i < 16; ++i) bf1[i] = q[i];
            } else {
#pragma unroll
                for (int i = 0; i < 16; ++i) bf1[i] = (f16)0.0f;
            }
        }
        const f16* p0 = wr0 + kk * 32;
        const f16* p1 = wr1 + kk * 32;
        v16h a0, a1;
#pragma unroll
        for (int i = 0; i < 8; ++i) {
            a0[i]     = p0[16 * half + i];     a0[8 + i] = p0[16 * half + 8 + i];
            a1[i]     = p1[16 * half + i];     a1[8 + i] = p1[16 * half + 8 + i];
        }
        acc00 = __builtin_amdgcn_wmma_f32_16x16x32_f16(false, a0, false, bf0, (short)0, acc00, false, false);
        acc01 = __builtin_amdgcn_wmma_f32_16x16x32_f16(false, a0, false, bf1, (short)0, acc01, false, false);
        acc10 = __builtin_amdgcn_wmma_f32_16x16x32_f16(false, a1, false, bf0, (short)0, acc10, false, false);
        acc11 = __builtin_amdgcn_wmma_f32_16x16x32_f16(false, a1, false, bf1, (short)0, acc11, false, false);
    }

    float lloss = 0.0f;
    const size_t pixa = ((size_t)(b * J_ + j) * L_ + la) * C_;
    const size_t pixb = ((size_t)(b * J_ + j) * L_ + lb) * C_;
#pragma unroll
    for (int r = 0; r < 8; ++r) {
        int ca = cout0 + r + 8 * half;
        int cb2 = cout1 + r + 8 * half;
        // (cout tile 0, N tile 0)
        {
            float y  = acc00[r] + pb[ca];
            float hb = 0.5f * (float)h16[pixa + ca] + 0.5f * y;
            size_t o = ((size_t)(b * C_ + ca) * J_ + j) * L_ + la;
            float fh = f_hat[o] + hb;
            f_hat[o]  = fh;  f_rest[o] -= hb;
            float d = fh - f[o];  lloss += d * d;
        }
        // (cout tile 0, N tile 1)
        {
            float y  = acc01[r] + pb[ca];
            float hb = 0.5f * (float)h16[pixb + ca] + 0.5f * y;
            size_t o = ((size_t)(b * C_ + ca) * J_ + j) * L_ + lb;
            float fh = f_hat[o] + hb;
            f_hat[o]  = fh;  f_rest[o] -= hb;
            float d = fh - f[o];  lloss += d * d;
        }
        // (cout tile 1, N tile 0)
        {
            float y  = acc10[r] + pb[cb2];
            float hb = 0.5f * (float)h16[pixa + cb2] + 0.5f * y;
            size_t o = ((size_t)(b * C_ + cb2) * J_ + j) * L_ + la;
            float fh = f_hat[o] + hb;
            f_hat[o]  = fh;  f_rest[o] -= hb;
            float d = fh - f[o];  lloss += d * d;
        }
        // (cout tile 1, N tile 1)
        {
            float y  = acc11[r] + pb[cb2];
            float hb = 0.5f * (float)h16[pixb + cb2] + 0.5f * y;
            size_t o = ((size_t)(b * C_ + cb2) * J_ + j) * L_ + lb;
            float fh = f_hat[o] + hb;
            f_hat[o]  = fh;  f_rest[o] -= hb;
            float d = fh - f[o];  lloss += d * d;
        }
    }
#pragma unroll
    for (int off = 16; off >= 1; off >>= 1) lloss += __shfl_xor(lloss, off, 32);
    if (lane == 0) s_loss[w] = lloss;
    __syncthreads();
    if (tid == 0) {
        float tot = 0.0f;
        for (int i = 0; i < 8; ++i) tot += s_loss[i];
        atomicAdd(loss_acc + si, tot);
    }
}

// --------------------------------------------------------- perplexity + loss
__global__ void k_count(const int* __restrict__ idx, float* __restrict__ counts, int n) {
    int i = blockIdx.x * blockDim.x + threadIdx.x;
    if (i < n) atomicAdd(&counts[idx[i]], 1.0f);
}

__global__ void __launch_bounds__(256) k_final(const float* __restrict__ counts,
                                               const float* __restrict__ loss_acc,
                                               float* __restrict__ out_loss,
                                               float* __restrict__ out_perp) {
    __shared__ float red[256];
    __shared__ float total;
    int tid = threadIdx.x;
    float s = 0.0f;
    for (int i = tid; i < NB; i += 256) s += counts[i];
    red[tid] = s; __syncthreads();
    for (int st = 128; st > 0; st >>= 1) { if (tid < st) red[tid] += red[tid + st]; __syncthreads(); }
    if (tid == 0) total = red[0];
    __syncthreads();
    float e = 0.0f;
    for (int i = tid; i < NB; i += 256) { float p = counts[i] / total; e -= p * logf(p + 1e-7f); }
    red[tid] = e; __syncthreads();
    for (int st = 128; st > 0; st >>= 1) { if (tid < st) red[tid] += red[tid + st]; __syncthreads(); }
    if (tid == 0) {
        *out_perp = expf(red[0]);
        float ls = 0.0f;
        for (int i = 0; i < 5; ++i) ls += loss_acc[i];
        *out_loss = ls / (float)NUMEL / 5.0f;   // BETA = 1
    }
}

// -------------------------------------------------------------------- launch
extern "C" void kernel_launch(void* const* d_in, const int* in_sizes, int n_in,
                              void* d_out, int out_size, void* d_ws, size_t ws_size,
                              hipStream_t stream) {
    (void)in_sizes; (void)n_in; (void)out_size; (void)ws_size;
    const float* f  = (const float*)d_in[0];
    const float* cb = (const float*)d_in[1];
    const float* pw = (const float*)d_in[2];
    const float* pb = (const float*)d_in[3];

    float* f_hat    = (float*)d_out;            // NUMEL, NCHW (== f_hat_out)
    float* out_loss = f_hat + NUMEL;
    float* out_perp = f_hat + NUMEL + 1;

    char* ws = (char*)d_ws;
    float* f_rest   = (float*)ws; ws += (size_t)NUMEL * 4;          // 32 MB
    f16*   h16      = (f16*)ws;   ws += (size_t)NUMEL * 2;          // 16 MB
    f16*   r16      = (f16*)ws;   ws += (size_t)32768 * 256 * 2;    // 16 MB
    f16*   cb16     = (f16*)ws;   ws += (size_t)NB * C_ * 2;        //  4 MB
    f16*   w16      = (f16*)ws;   ws += (size_t)4 * C_ * K9 * 2;    // 4.5 MB
    float* cb_sq    = (float*)ws; ws += (size_t)NB * 4;
    float* counts   = (float*)ws; ws += (size_t)NB * 4;
    float* loss_acc = (float*)ws; ws += 8 * 4;
    int*   idx_buf  = (int*)ws;

    k_init<<<NUMEL / 256, 256, 0, stream>>>(f, f_rest, f_hat);
    k_init_small<<<NB / 256, 256, 0, stream>>>(counts, loss_acc);
    k_prep_cb<<<NB * C_ / 256, 256, 0, stream>>>(cb, cb16, cb_sq);
    k_prep_w<<<(4 * C_ * K9 + 255) / 256, 256, 0, stream>>>(pw, w16);

    const int SN[5]  = {1, 2, 4, 8, 16};
    const int TN[5]  = {4, 16, 64, 128, 256};
    const int PHI[5] = {0, 1, 1, 2, 3};
    int off = 0;
    for (int si = 0; si < 5; ++si) {
        int sn = SN[si], tn = TN[si], N = B_ * sn * tn;
        k_area<<<(N * C_) / 256, 256, 0, stream>>>(f_rest, r16, sn, tn);
        k_argmin<<<N / 16, 256, 0, stream>>>(r16, cb16, cb_sq, idx_buf + off);
        k_upsample<<<NUMEL / 256, 256, 0, stream>>>(cb, idx_buf + off, h16, sn, tn);
        k_conv<<<1024, 256, 0, stream>>>(h16, w16 + (size_t)PHI[si] * C_ * K9,
                                         pb + PHI[si] * C_, f, f_hat, f_rest, loss_acc, si);
        off += N;
    }
    k_count<<<(NIDX + 255) / 256, 256, 0, stream>>>(idx_buf, counts, NIDX);
    k_final<<<1, 256, 0, stream>>>(counts, loss_acc, out_loss, out_perp);
}
